// TSP_GGCN_12678743458067
// MI455X (gfx1250) — compile-verified, matrix operand order checked by reading, not run on hardware
//
#include <hip/hip_runtime.h>
#include <hip/hip_bf16.h>
#include <math.h>

// ---------------------------------------------------------------------------
// GatedGraphConv x2 + edge MLP for MI455X (gfx1250, wave32).
// GEMMs: V_WMMA_F32_16X16X32_F16, operands pre-packed to zero-padded f16
// row-major so each fragment is exactly 2x global_load_b128 per lane
// (ISA 7.12.2 16-bit A layout == contiguous row segments).
// ---------------------------------------------------------------------------

typedef __attribute__((ext_vector_type(16))) _Float16 v16h;
typedef __attribute__((ext_vector_type(8)))  _Float16 v8h;
typedef __attribute__((ext_vector_type(8)))  float    v8f;

#define XS  112   // f32 node-feature stride (>=100, mult of 16)
#define GS  304   // f32 gate stride (>=300, mult of 16)
#define KAP 128   // f16 packed K stride (>=100, mult of 32)

// ---- elementwise / packing helpers -----------------------------------------

// x[n, 0:2] = nf, rest 0; also emits f16 packed copy.
__global__ void pad_input_kernel(const float* __restrict__ nf,
                                 float* __restrict__ x,
                                 _Float16* __restrict__ xh, int N) {
    int idx = blockIdx.x * blockDim.x + threadIdx.x;
    if (idx >= N * KAP) return;
    int n = idx / KAP, k = idx % KAP;
    float v = (k < 2) ? nf[n * 2 + k] : 0.0f;
    if (k < XS) x[(size_t)n * XS + k] = v;
    xh[idx] = (_Float16)v;
}

__global__ void zero_kernel(float* __restrict__ p, int n) {
    int idx = blockIdx.x * blockDim.x + threadIdx.x;
    if (idx < n) p[idx] = 0.0f;
}

// relu live channels, zero padding; refresh f16 packed copy.
__global__ void relu_expand_kernel(float* __restrict__ x,
                                   _Float16* __restrict__ xh, int N, int C) {
    int idx = blockIdx.x * blockDim.x + threadIdx.x;
    if (idx >= N * KAP) return;
    int n = idx / KAP, k = idx % KAP;
    float v = 0.0f;
    if (k < C) {
        v = fmaxf(x[(size_t)n * XS + k], 0.0f);
        x[(size_t)n * XS + k] = v;
    } else if (k < XS) {
        x[(size_t)n * XS + k] = 0.0f;
    }
    xh[idx] = (_Float16)v;
}

// f32 [N x lda] -> zero-padded f16 [N x KAP]
__global__ void pack_a_kernel(const float* __restrict__ A, int lda,
                              _Float16* __restrict__ Ah, int N, int C) {
    int idx = blockIdx.x * blockDim.x + threadIdx.x;
    if (idx >= N * KAP) return;
    int n = idx / KAP, k = idx % KAP;
    Ah[idx] = (_Float16)((k < C) ? A[(size_t)n * lda + k] : 0.0f);
}

// Pack weights into B^T-row-major f16 [rowsP x KAP].
// trans==0: src already [n,k] row-major ([srcR x srcC], e.g. wih)
// trans==1: src is [k,n] row-major ([srcR x srcC]), take transpose (e.g. W, fcn_w)
__global__ void pack_b_kernel(const float* __restrict__ src, int srcR, int srcC,
                              int trans, _Float16* __restrict__ Bh, int rowsP) {
    int idx = blockIdx.x * blockDim.x + threadIdx.x;
    if (idx >= rowsP * KAP) return;
    int n = idx / KAP, k = idx % KAP;
    float v = 0.0f;
    if (trans) {                       // Bh[n,k] = src[k,n]
        if (k < srcR && n < srcC) v = src[(size_t)k * srcC + n];
    } else {                           // Bh[n,k] = src[n,k]
        if (n < srcR && k < srcC) v = src[(size_t)n * srcC + k];
    }
    Bh[idx] = (_Float16)v;
}

// ---- WMMA GEMM: Out[M x tilesK*16] = Ah[M x KAP] * Bh^T (+bias)(+relu) -----
// One wave32 per 16x16 tile; tile choice wave-uniform => EXEC all-ones at WMMA.
// Fragment = 2 x b128 per lane straight from packed f16 rows (no guards).
__global__ void gemm_wmma_kernel(const _Float16* __restrict__ Ah,
                                 const _Float16* __restrict__ Bh,
                                 const float* __restrict__ bias,
                                 float* __restrict__ Out, int ldo,
                                 int M, int Kout, int tilesK, int kBlocks,
                                 int relu) {
    int wave    = blockIdx.x * (blockDim.x >> 5) + (threadIdx.x >> 5);
    int lane    = threadIdx.x & 31;
    int tileRow = wave / tilesK;            // tileCol fastest: waves in a block
    int tileCol = wave % tilesK;            // share the A row tile (L0 reuse)
    int mBase   = tileRow * 16;
    if (mBase >= M) return;                 // wave-uniform exit

    int half = lane >> 4;
    int l16  = lane & 15;

    // ISA 7.12.2: lane's VGPR0-3 = row (lane&15), K = half*8..half*8+7 (16B),
    // VGPR4-7 = same +16 halves. Both 16B-aligned.
    const _Float16* Ap = Ah + (size_t)(mBase + l16) * KAP + half * 8;
    const _Float16* Bp = Bh + (size_t)(tileCol * 16 + l16) * KAP + half * 8;

    v8f acc = {};
    for (int kb = 0; kb < kBlocks; ++kb) {
        v8h alo = *(const v8h*)(Ap);
        v8h ahi = *(const v8h*)(Ap + 16);
        v8h blo = *(const v8h*)(Bp);
        v8h bhi = *(const v8h*)(Bp + 16);
        v16h a = __builtin_shufflevector(alo, ahi, 0, 1, 2, 3, 4, 5, 6, 7,
                                         8, 9, 10, 11, 12, 13, 14, 15);
        v16h b = __builtin_shufflevector(blo, bhi, 0, 1, 2, 3, 4, 5, 6, 7,
                                         8, 9, 10, 11, 12, 13, 14, 15);
        acc = __builtin_amdgcn_wmma_f32_16x16x32_f16(false, a, false, b,
                                                     (short)0, acc, false, false);
        Ap += 32;
        Bp += 32;
    }

    // C/D layout: N = lane&15; M = vgpr + 8*half
#pragma unroll
    for (int v = 0; v < 8; ++v) {
        int mm = mBase + v + 8 * half;
        int nn = tileCol * 16 + l16;
        if (mm < M) {
            float val = 0.0f;                       // zero pad columns
            if (nn < Kout) {
                val = acc[v];
                if (bias) val += bias[nn];
                if (relu) val = fmaxf(val, 0.0f);
            }
            Out[(size_t)mm * ldo + nn] = val;
        }
    }
}

// ---- edge scatter: agg[dst] += w_e * m[src] (L2-resident atomics) ----------
__global__ void scatter_kernel(const float* __restrict__ m,
                               const int* __restrict__ src,
                               const int* __restrict__ dst,
                               const float* __restrict__ w,
                               float* __restrict__ agg,
                               long long total, int C) {
    long long idx = (long long)blockIdx.x * blockDim.x + threadIdx.x;
    if (idx >= total) return;
    int e = (int)(idx / C);
    int c = (int)(idx % C);
    int s = src[e], d = dst[e];
    atomicAdd(&agg[(size_t)d * XS + c], m[(size_t)s * XS + c] * w[e]);
}

// ---- GRUCell elementwise (gates r, z, n); also refreshes f16 x -------------
__global__ void gru_kernel(const float* __restrict__ gi,
                           const float* __restrict__ gh,
                           float* __restrict__ x,
                           _Float16* __restrict__ xh, int N, int C) {
    int idx = blockIdx.x * blockDim.x + threadIdx.x;
    if (idx >= N * C) return;
    int n = idx / C, c = idx % C;
    const float* gin = gi + (size_t)n * GS;
    const float* ghn = gh + (size_t)n * GS;
    float ir = gin[c], iz = gin[C + c], inn = gin[2 * C + c];
    float hr = ghn[c], hz = ghn[C + c], hn  = ghn[2 * C + c];
    float r  = 1.0f / (1.0f + expf(-(ir + hr)));
    float z  = 1.0f / (1.0f + expf(-(iz + hz)));
    float nt = tanhf(inn + r * hn);
    float h  = x[(size_t)n * XS + c];
    float o  = (1.0f - z) * nt + z * h;
    x[(size_t)n * XS + c]  = o;
    xh[(size_t)n * KAP + c] = (_Float16)o;
}

// ---- final edge head: out[e] = [y[src], y[dst]] @ fce_w + fce_b ------------
__global__ void edge_out_kernel(const float* __restrict__ y,
                                const int* __restrict__ src,
                                const int* __restrict__ dst,
                                const float* __restrict__ fw,
                                const float* __restrict__ fb,
                                float* __restrict__ out, int E) {
    int e = blockIdx.x * blockDim.x + threadIdx.x;
    if (e >= E) return;
    int s = src[e], d = dst[e];
    const float* ys = y + (size_t)s * XS;
    const float* yd = y + (size_t)d * XS;
    float a0 = fb[0], a1 = fb[1];
#pragma unroll 4
    for (int k = 0; k < 100; ++k) {
        float v = ys[k];
        a0 += v * fw[2 * k];
        a1 += v * fw[2 * k + 1];
    }
#pragma unroll 4
    for (int k = 0; k < 100; ++k) {
        float v = yd[k];
        a0 += v * fw[2 * (100 + k)];
        a1 += v * fw[2 * (100 + k) + 1];
    }
    out[2 * e]     = a0;
    out[2 * e + 1] = a1;
}

// ---------------------------------------------------------------------------

extern "C" void kernel_launch(void* const* d_in, const int* in_sizes, int n_in,
                              void* d_out, int out_size, void* d_ws, size_t ws_size,
                              hipStream_t stream) {
    const float* nf   = (const float*)d_in[0];
    const int*   ei   = (const int*)d_in[1];
    const float* dist = (const float*)d_in[2];
    const float* w1   = (const float*)d_in[3];
    const float* wih1 = (const float*)d_in[4];
    const float* whh1 = (const float*)d_in[5];
    const float* bih1 = (const float*)d_in[6];
    const float* bhh1 = (const float*)d_in[7];
    const float* w2   = (const float*)d_in[8];
    const float* wih2 = (const float*)d_in[9];
    const float* whh2 = (const float*)d_in[10];
    const float* bih2 = (const float*)d_in[11];
    const float* bhh2 = (const float*)d_in[12];
    const float* fcnw = (const float*)d_in[13];
    const float* fcnb = (const float*)d_in[14];
    const float* fcew = (const float*)d_in[15];
    const float* fceb = (const float*)d_in[16];

    const int N = in_sizes[0] / 2;
    const int E = in_sizes[2];
    const int* src = ei;
    const int* dst = ei + E;

    // ---- workspace carve-up -------------------------------------------------
    float* x   = (float*)d_ws;                 // N*XS
    float* m   = x   + (size_t)N * XS;         // N*XS (also final y)
    float* agg = m   + (size_t)N * XS;         // N*XS
    float* gi  = agg + (size_t)N * XS;         // N*GS
    float* gh  = gi  + (size_t)N * GS;         // N*GS
    _Float16* xh   = (_Float16*)(gh + (size_t)N * GS);  // N*KAP
    _Float16* aggh = xh + (size_t)N * KAP;               // N*KAP
    _Float16* wp   = aggh + (size_t)N * KAP;             // packed weights
    _Float16* wh1[2] = {wp, wp + 64 * KAP};
    _Float16* wih1h  = wh1[1] + 64 * KAP;
    _Float16* whh1h  = wih1h + 160 * KAP;
    _Float16* wh2[3] = {whh1h + 160 * KAP, whh1h + (160 + 112) * KAP,
                        whh1h + (160 + 224) * KAP};
    _Float16* wih2h  = wh2[2] + 112 * KAP;
    _Float16* whh2h  = wih2h + 304 * KAP;
    _Float16* fcnh   = whh2h + 304 * KAP;      // 112*KAP

    const int TPB = 256;  // 8 waves per block (wave32)
    auto nblk = [](long long n, int tpb) { return (int)((n + tpb - 1) / tpb); };
    const int rowTiles = (N + 15) / 16;

    // ---- pack all weights to f16 (tiny) ------------------------------------
    for (int l = 0; l < 2; ++l)
        pack_b_kernel<<<nblk(64 * KAP, TPB), TPB, 0, stream>>>(
            w1 + (size_t)l * 50 * 50, 50, 50, /*trans=*/1, wh1[l], 64);
    pack_b_kernel<<<nblk(160 * KAP, TPB), TPB, 0, stream>>>(wih1, 150, 50, 0, wih1h, 160);
    pack_b_kernel<<<nblk(160 * KAP, TPB), TPB, 0, stream>>>(whh1, 150, 50, 0, whh1h, 160);
    for (int l = 0; l < 3; ++l)
        pack_b_kernel<<<nblk(112 * KAP, TPB), TPB, 0, stream>>>(
            w2 + (size_t)l * 100 * 100, 100, 100, 1, wh2[l], 112);
    pack_b_kernel<<<nblk(304 * KAP, TPB), TPB, 0, stream>>>(wih2, 300, 100, 0, wih2h, 304);
    pack_b_kernel<<<nblk(304 * KAP, TPB), TPB, 0, stream>>>(whh2, 300, 100, 0, whh2h, 304);
    pack_b_kernel<<<nblk(112 * KAP, TPB), TPB, 0, stream>>>(fcnw, 100, 100, 1, fcnh, 112);

    pad_input_kernel<<<nblk((long long)N * KAP, TPB), TPB, 0, stream>>>(nf, x, xh, N);

    // ---- one GGC iteration --------------------------------------------------
    auto ggc_iter = [&](const _Float16* Wl, const _Float16* wihh,
                        const _Float16* whhh, const float* bih,
                        const float* bhh, int C) {
        const int ctK     = (C + 15) / 16;
        const int gtK     = (3 * C + 15) / 16;
        const int kBlocks = (C + 31) / 32;

        // m = x @ W[l]
        {
            long long waves = (long long)rowTiles * ctK;
            gemm_wmma_kernel<<<nblk(waves * 32, TPB), TPB, 0, stream>>>(
                xh, Wl, nullptr, m, XS, N, C, ctK, kBlocks, 0);
        }
        // agg = 0 ; agg[dst] += dist * m[src]
        zero_kernel<<<nblk((long long)N * XS, TPB), TPB, 0, stream>>>(agg, N * XS);
        {
            long long total = (long long)E * C;
            scatter_kernel<<<nblk(total, TPB), TPB, 0, stream>>>(
                m, src, dst, dist, agg, total, C);
        }
        // pack agg -> f16
        pack_a_kernel<<<nblk((long long)N * KAP, TPB), TPB, 0, stream>>>(
            agg, XS, aggh, N, C);
        // gi = agg @ wih^T + bih ; gh = x @ whh^T + bhh
        {
            long long waves = (long long)rowTiles * gtK;
            gemm_wmma_kernel<<<nblk(waves * 32, TPB), TPB, 0, stream>>>(
                aggh, wihh, bih, gi, GS, N, 3 * C, gtK, kBlocks, 0);
            gemm_wmma_kernel<<<nblk(waves * 32, TPB), TPB, 0, stream>>>(
                xh, whhh, bhh, gh, GS, N, 3 * C, gtK, kBlocks, 0);
        }
        // x = GRUCell(agg, x)  (also refresh xh)
        gru_kernel<<<nblk((long long)N * C, TPB), TPB, 0, stream>>>(
            gi, gh, x, xh, N, C);
    };

    // GGC stage 1: 50 channels, 2 layers
    for (int l = 0; l < 2; ++l)
        ggc_iter(wh1[l], wih1h, whh1h, bih1, bhh1, 50);
    relu_expand_kernel<<<nblk((long long)N * KAP, TPB), TPB, 0, stream>>>(x, xh, N, 50);

    // GGC stage 2: 100 channels, 3 layers
    for (int l = 0; l < 3; ++l)
        ggc_iter(wh2[l], wih2h, whh2h, bih2, bhh2, 100);
    relu_expand_kernel<<<nblk((long long)N * KAP, TPB), TPB, 0, stream>>>(x, xh, N, 100);

    // y = relu(x @ fcn_w + fcn_b)  -> reuse m as y
    {
        const int ctK = 7;  // 112 cols
        long long waves = (long long)rowTiles * ctK;
        gemm_wmma_kernel<<<nblk(waves * 32, TPB), TPB, 0, stream>>>(
            xh, fcnh, fcnb, m, XS, N, 100, ctK, 4, 1);
    }

    // edge head
    edge_out_kernel<<<nblk((long long)E, TPB), TPB, 0, stream>>>(
        m, src, dst, fcew, fceb, (float*)d_out, E);
}
